// CrossIQ_30511447670998
// MI455X (gfx1250) — matrix operand order repaired and checked by reading, not compiled, not run on hardware
//
#include <hip/hip_runtime.h>

typedef __attribute__((ext_vector_type(16))) __bf16 v16bf;
typedef __attribute__((ext_vector_type(8)))  float  v8f;

union FragB16 { unsigned u[8]; v16bf v; };

__device__ __forceinline__ unsigned short f2bf(float f) {
  unsigned u = __builtin_bit_cast(unsigned, f);
  u += 0x7FFFu + ((u >> 16) & 1u);          // round-to-nearest-even
  return (unsigned short)(u >> 16);
}
__device__ __forceinline__ unsigned pack2bf(float lo, float hi) {
  return (unsigned)f2bf(lo) | ((unsigned)f2bf(hi) << 16);
}
__device__ __forceinline__ float bf2f(unsigned short h) {
  unsigned u = ((unsigned)h) << 16;
  return __builtin_bit_cast(float, u);
}

// ---- CDNA5 async global->LDS copy (ASYNCcnt-tracked, no VGPR data path) ----
// VDST VGPR carries the per-lane LDS byte offset (LDS_BASE-relative); dynamic
// LDS starts at offset 0 in kernels with no static __shared__, so plain byte
// offsets into the extern smem block are valid.
__device__ __forceinline__ void async_g2l_b128(const void* gaddr, unsigned lds_off) {
  asm volatile("global_load_async_to_lds_b128 %0, %1, off"
               :: "v"(lds_off), "v"(gaddr) : "memory");
}
__device__ __forceinline__ void async_wait0() {
  asm volatile("s_wait_asynccnt 0x0" ::: "memory");
}

enum { MODE_BF16_PM = 0, MODE_BF16_CM = 1, MODE_GATE = 2, MODE_PLAIN = 3 };

// ---------------------------------------------------------------------------
// Generic 1x1-conv GEMM:  Y[M=Cout][N=pixels] = W[Cout][Cin] * X[Cin][N]
// 256 threads = 8 waves; macro tile 64(M) x 128(N); K step 32 (one WMMA deep)
// (staging converts f32->bf16, so it must go through VGPRs; no async here)
// ---------------------------------------------------------------------------
__global__ __launch_bounds__(256) void conv1x1_wmma(
    const float* __restrict__ X, const float* __restrict__ W,
    const float* __restrict__ bias, const float* __restrict__ sptr, int sidx,
    float scale_const, int mode, float beta, const float* __restrict__ T,
    void* __restrict__ outp, int Cin, int Cout, int Npix)
{
  __shared__ unsigned lds_w[64 * 17];    // 64 M rows x 16 k-pairs (padded)
  __shared__ unsigned lds_x[16 * 132];   // 16 k-pair rows x 128 n (padded)

  const int tid  = threadIdx.x;
  const int lane = tid & 31;
  const int wv   = tid >> 5;
  const int wrow = wv >> 2;    // 0..1  (M direction)
  const int wcol = wv & 3;     // 0..3  (N direction)
  const int hi   = lane >> 4;  // lane half
  const int l15  = lane & 15;

  const int m0 = blockIdx.x * 64;
  const int n0 = blockIdx.y * 128;
  const int z  = blockIdx.z;
  X += (size_t)z * Cin * Npix;

  v8f acc[2][2];
#pragma unroll
  for (int a = 0; a < 2; a++)
#pragma unroll
    for (int b = 0; b < 2; b++)
#pragma unroll
      for (int v = 0; v < 8; v++) acc[a][b][v] = 0.0f;

  for (int k0 = 0; k0 < Cin; k0 += 32) {
    __syncthreads();
#pragma unroll
    for (int i = 0; i < 4; i++) {
      int idx = tid + i * 256;
      int m = idx >> 4, kp = idx & 15;
      const float2 wf = *(const float2*)(W + (size_t)(m0 + m) * Cin + k0 + 2 * kp);
      lds_w[m * 17 + kp] = pack2bf(wf.x, wf.y);
    }
#pragma unroll
    for (int i = 0; i < 8; i++) {
      int idx = tid + i * 256;
      int n = idx & 127, kp = idx >> 7;
      const float* xp = X + (size_t)(k0 + 2 * kp) * Npix + n0 + n;
      lds_x[kp * 132 + n] = pack2bf(xp[0], xp[Npix]);
    }
    if (k0 + 32 < Cin)
      __builtin_prefetch(X + (size_t)(k0 + 32) * Npix + n0 + (tid & 127), 0, 3);
    __syncthreads();

    FragB16 af[2], bf[2];
#pragma unroll
    for (int mt = 0; mt < 2; mt++) {
      int M = wrow * 32 + mt * 16 + l15;
#pragma unroll
      for (int v = 0; v < 8; v++) {
        int kk = (v & 3) * 2 + ((v >> 2) << 4) + (hi << 3);   // A-frag K pattern
        af[mt].u[v] = lds_w[M * 17 + (kk >> 1)];
      }
    }
#pragma unroll
    for (int nt = 0; nt < 2; nt++) {
      int nn = wcol * 32 + nt * 16 + l15;
#pragma unroll
      for (int v = 0; v < 8; v++) {
        int kp = (v & 3) + ((v >> 2) << 3) + (hi << 2);       // B-frag K pattern
        bf[nt].u[v] = lds_x[kp * 132 + nn];
      }
    }
#pragma unroll
    for (int mt = 0; mt < 2; mt++)
#pragma unroll
      for (int nt = 0; nt < 2; nt++)
        acc[mt][nt] = __builtin_amdgcn_wmma_f32_16x16x32_bf16(
            false, af[mt].v, false, bf[nt].v, (short)0, acc[mt][nt], false, false);
  }

  const float scale = scale_const * (sidx >= 0 ? sptr[sidx] : 1.0f);
#pragma unroll
  for (int mt = 0; mt < 2; mt++) {
#pragma unroll
    for (int nt = 0; nt < 2; nt++) {
#pragma unroll
      for (int v = 0; v < 8; v++) {
        int Mg = m0 + wrow * 32 + mt * 16 + v + hi * 8;
        int Ng = n0 + wcol * 32 + nt * 16 + l15;
        float val = acc[mt][nt][v] + bias[Mg];
        if (mode == MODE_BF16_PM) {          // pixel-major bf16 (Q, K)
          unsigned short* o = (unsigned short*)outp + (size_t)z * Npix * Cout;
          o[(size_t)Ng * Cout + Mg] = f2bf(val * scale);
        } else if (mode == MODE_BF16_CM) {   // channel-major bf16 (V)
          unsigned short* o = (unsigned short*)outp + (size_t)z * Cout * Npix;
          o[(size_t)Mg * Npix + Ng] = f2bf(val * scale);
        } else if (mode == MODE_GATE) {      // fused += sigmoid(conv)*t
          const float* tp = T + (size_t)z * Cout * Npix;
          float g = 1.0f / (1.0f + __expf(-val));
          float r = g * tp[(size_t)Mg * Npix + Ng];
          float* o = (float*)outp + (size_t)z * Cout * Npix;
          size_t oi = (size_t)Mg * Npix + Ng;
          o[oi] = beta * o[oi] + r;
        } else {                             // plain f32 (final conv)
          float* o = (float*)outp + (size_t)z * Cout * Npix;
          o[(size_t)Mg * Npix + Ng] = val;
        }
      }
    }
  }
}

// ---------------------------------------------------------------------------
// Ksum: KS[m] = K[pa[m]] + K[pb[m]]  (bf16 pairs processed as dwords)
// ---------------------------------------------------------------------------
__global__ __launch_bounds__(256) void ksum_bf16(const unsigned* __restrict__ K,
                                                 unsigned* __restrict__ KS, int ndw)
{
  const int pa[3] = {2, 0, 0};   // rgb: kd+kt ; depth: kr+kt ; thermal: kr+kd
  const int pb[3] = {1, 1, 2};
  const int m = blockIdx.y;
  const unsigned* A  = K + (size_t)pa[m] * ndw;
  const unsigned* Bp = K + (size_t)pb[m] * ndw;
  unsigned* O = KS + (size_t)m * ndw;
  for (int i = blockIdx.x * blockDim.x + threadIdx.x; i < ndw;
       i += gridDim.x * blockDim.x) {
    unsigned a = A[i], b = Bp[i];
    O[i] = pack2bf(bf2f((unsigned short)(a & 0xffffu)) + bf2f((unsigned short)(b & 0xffffu)),
                   bf2f((unsigned short)(a >> 16))     + bf2f((unsigned short)(b >> 16)));
  }
}

// ---------------------------------------------------------------------------
// Flash attention with double-buffered async global->LDS staging.
// 128 threads = 4 waves; wave w owns queries [w*16,w*16+16) and all 256
// output channels (16 accumulator tiles = 128 VGPRs).
// LDS byte map (dynamic smem, base offset 0):
//   [0,33792)          Q tile   : 64 rows x 33 uint4 (pitch 132 dw)
//   [33792,101376)     Ksum x2  : each 64 x 33 uint4
//   [101376,175104)    V x2     : each 256 c-rows x 36 dw (B128-aligned pitch)
//   [175104,183808)    P        : 64 rows x 34 dw (68 ushorts)
// ---------------------------------------------------------------------------
#define OFF_Q   0u
#define OFF_KS  33792u
#define SZ_KS   33792u
#define OFF_V   101376u
#define SZ_V    36864u
#define OFF_P   175104u
#define SMEM_BYTES 183808u

__global__ __launch_bounds__(128) void flashattn_wmma(
    const unsigned* __restrict__ Qg,   // [B][N][C/2] dwords (bf16 pairs)
    const unsigned* __restrict__ KSg,  // [B][N][C/2]
    const unsigned* __restrict__ Vg,   // [B][C][N/2]
    float* __restrict__ Og,            // [B][C][N]
    int Npix)
{
  extern __shared__ char smem[];
  unsigned* qd  = (unsigned*)(smem + OFF_Q);            // pitch 132 dw
  unsigned* ks0 = (unsigned*)(smem + OFF_KS);
  unsigned* ks1 = (unsigned*)(smem + OFF_KS + SZ_KS);
  unsigned* vv0 = (unsigned*)(smem + OFF_V);
  unsigned* vv1 = (unsigned*)(smem + OFF_V + SZ_V);
  unsigned* pd  = (unsigned*)(smem + OFF_P);
  unsigned short* pus = (unsigned short*)pd;

  const int tid  = threadIdx.x;
  const int lane = tid & 31;
  const int w    = tid >> 5;
  const int hi   = lane >> 4;
  const int l15  = lane & 15;

  const int n0 = blockIdx.x * 64;
  const int b  = blockIdx.y;
  Qg  += (size_t)b * Npix * 128;
  KSg += (size_t)b * Npix * 128;
  Vg  += (size_t)b * 256 * (Npix >> 1);
  Og  += (size_t)b * 256 * Npix;

  const uint4* qg4  = (const uint4*)Qg;
  const uint4* ksg4 = (const uint4*)KSg;

  // --- async-issue Q tile (once) ---
#pragma unroll
  for (int i = 0; i < 16; i++) {
    int idx = tid + i * 128;
    int row = idx >> 5, col = idx & 31;
    async_g2l_b128(qg4 + n0 * 32 + idx, OFF_Q + (unsigned)(row * 33 + col) * 16u);
  }
  // --- async-issue first Ksum + V tiles into buffer 0 ---
  {
    const int m0 = 0;
#pragma unroll
    for (int i = 0; i < 16; i++) {                 // Ksum: 16 x B128 per lane
      int idx = tid + i * 128;
      int row = idx >> 5, col = idx & 31;
      async_g2l_b128(ksg4 + m0 * 32 + idx, OFF_KS + (unsigned)(row * 33 + col) * 16u);
    }
    const int jj = tid & 7, cb = tid >> 3;
#pragma unroll
    for (int i = 0; i < 16; i++) {                 // V: 16 x B128 per lane
      int c = cb + 16 * i;
      async_g2l_b128(Vg + (size_t)c * (Npix >> 1) + (m0 >> 1) + 4 * jj,
                     OFF_V + (unsigned)(c * 36 + 4 * jj) * 4u);
    }
  }

  float rm[8], rs[8];
  v8f acc[16];
#pragma unroll
  for (int v = 0; v < 8; v++) { rm[v] = -3.0e38f; rs[v] = 0.0f; }
#pragma unroll
  for (int ct = 0; ct < 16; ct++)
#pragma unroll
    for (int v = 0; v < 8; v++) acc[ct][v] = 0.0f;

  const int ntiles = Npix >> 6;
  for (int it = 0; it < ntiles; ++it) {
    const int buf = it & 1;
    unsigned* ksd = buf ? ks1 : ks0;
    unsigned* vd  = buf ? vv1 : vv0;

    async_wait0();        // this tile's (and Q's) async copies landed in LDS
    __syncthreads();      // all waves waited; prev compute done -> other buf free

    if (it + 1 < ntiles) {          // prefetch next tile into the other buffer
      const int m1 = (it + 1) << 6;
      const unsigned oks = OFF_KS + (buf ? 0u : SZ_KS);
      const unsigned ov  = OFF_V  + (buf ? 0u : SZ_V);
#pragma unroll
      for (int i = 0; i < 16; i++) {
        int idx = tid + i * 128;
        int row = idx >> 5, col = idx & 31;
        async_g2l_b128(ksg4 + m1 * 32 + idx, oks + (unsigned)(row * 33 + col) * 16u);
      }
      const int jj = tid & 7, cb = tid >> 3;
#pragma unroll
      for (int i = 0; i < 16; i++) {
        int c = cb + 16 * i;
        async_g2l_b128(Vg + (size_t)c * (Npix >> 1) + (m1 >> 1) + 4 * jj,
                       ov + (unsigned)(c * 36 + 4 * jj) * 4u);
      }
    }

    // ---- scores S[16 q][64 m], K-depth 256 ----
    v8f s[4];
#pragma unroll
    for (int mt = 0; mt < 4; mt++)
#pragma unroll
      for (int v = 0; v < 8; v++) s[mt][v] = 0.0f;

    const int qrow = w * 16 + l15;
#pragma unroll
    for (int kc = 0; kc < 8; kc++) {
      FragB16 a;
#pragma unroll
      for (int v = 0; v < 8; v++) {
        int kk = kc * 32 + (v & 3) * 2 + ((v >> 2) << 4) + (hi << 3);
        a.u[v] = qd[qrow * 132 + (kk >> 1)];
      }
#pragma unroll
      for (int mt = 0; mt < 4; mt++) {
        FragB16 bm;
        int mcol = mt * 16 + l15;
#pragma unroll
        for (int v = 0; v < 8; v++) {
          int kk = kc * 32 + (v & 3) * 2 + ((v >> 2) << 4) + (hi << 3);
          bm.u[v] = ksd[mcol * 132 + (kk >> 1)];
        }
        s[mt] = __builtin_amdgcn_wmma_f32_16x16x32_bf16(
            false, a.v, false, bm.v, (short)0, s[mt], false, false);
      }
    }

    // ---- online softmax (rows in (lane-half, vgpr); cols across 16 lanes) ---
#pragma unroll
    for (int v = 0; v < 8; v++) {
      float m = rm[v];
#pragma unroll
      for (int mt = 0; mt < 4; mt++) m = fmaxf(m, s[mt][v]);
      m = fmaxf(m, __shfl_xor(m, 1, 32));
      m = fmaxf(m, __shfl_xor(m, 2, 32));
      m = fmaxf(m, __shfl_xor(m, 4, 32));
      m = fmaxf(m, __shfl_xor(m, 8, 32));
      float corr = __expf(rm[v] - m);
      rm[v] = m;
      float sum = 0.0f;
#pragma unroll
      for (int mt = 0; mt < 4; mt++) {
        float p = __expf(s[mt][v] - m);
        s[mt][v] = p;
        sum += p;
      }
      sum += __shfl_xor(sum, 1, 32);
      sum += __shfl_xor(sum, 2, 32);
      sum += __shfl_xor(sum, 4, 32);
      sum += __shfl_xor(sum, 8, 32);
      rs[v] = rs[v] * corr + sum;
#pragma unroll
      for (int ct = 0; ct < 16; ct++) acc[ct][v] *= corr;
    }

    // ---- P -> LDS (per-wave region, D-layout -> A-layout repack) ----
#pragma unroll
    for (int mt = 0; mt < 4; mt++)
#pragma unroll
      for (int v = 0; v < 8; v++)
        pus[(w * 16 + v + hi * 8) * 68 + mt * 16 + l15] = f2bf(s[mt][v]);

    // ---- O += P[16x32] * V^T[32x16] over all 16 channel tiles ----
#pragma unroll
    for (int kt = 0; kt < 2; kt++) {
      FragB16 pa;
#pragma unroll
      for (int v = 0; v < 8; v++) {
        int kk = kt * 32 + (v & 3) * 2 + ((v >> 2) << 4) + (hi << 3);
        pa.u[v] = pd[(w * 16 + l15) * 34 + (kk >> 1)];
      }
#pragma unroll
      for (int ct = 0; ct < 16; ct++) {
        FragB16 vb;
        int c = ct * 16 + l15;
#pragma unroll
        for (int v = 0; v < 8; v++) {
          int kk = kt * 32 + (v & 3) * 2 + ((v >> 2) << 4) + (hi << 3);
          vb.u[v] = vd[c * 36 + (kk >> 1)];
        }
        acc[ct] = __builtin_amdgcn_wmma_f32_16x16x32_bf16(
            false, pa.v, false, vb.v, (short)0, acc[ct], false, false);
      }
    }
  }

  float inv[8];
#pragma unroll
  for (int v = 0; v < 8; v++) inv[v] = 1.0f / rs[v];
#pragma unroll
  for (int ct = 0; ct < 16; ct++) {
#pragma unroll
    for (int v = 0; v < 8; v++) {
      int c = ct * 16 + l15;
      int n = n0 + w * 16 + v + hi * 8;
      Og[(size_t)c * Npix + n] = acc[ct][v] * inv[v];
    }
  }
}

// ---------------------------------------------------------------------------
extern "C" void kernel_launch(void* const* d_in, const int* in_sizes, int n_in,
                              void* d_out, int out_size, void* d_ws, size_t ws_size,
                              hipStream_t stream)
{
  (void)in_sizes; (void)n_in; (void)out_size; (void)ws_size;
  const int Bb = 4, C = 256, HW = 48 * 48, OUT_C = 256;
  const size_t NB = (size_t)Bb * HW * C;     // elements per modality tensor

  // workspace carve: Q,K,KS,V bf16 [3][...], ATT & FUSED f32  (~75.5 MB total)
  unsigned short* Q  = (unsigned short*)d_ws;   // [3][B][N][C] pixel-major
  unsigned short* K  = Q  + 3 * NB;             // [3][B][N][C] pixel-major
  unsigned short* KS = K  + 3 * NB;             // [3][B][N][C]
  unsigned short* V  = KS + 3 * NB;             // [3][B][C][N] channel-major
  float* ATT   = (float*)(V + 3 * NB);          // [B][C][N]
  float* FUSED = ATT + NB;                      // [B][C][N]

  const float* s = (const float*)d_in[3];

  dim3 blk(256);
  dim3 grd(C / 64, HW / 128, Bb);

  // 1) nine QKV conv GEMMs. q gets s[i]/sqrt(C) folded in, k gets s[i].
  for (int mod = 0; mod < 3; mod++) {
    const float* x  = (const float*)d_in[mod];
    const float* wq = (const float*)d_in[4 + mod * 6 + 0];
    const float* bq = (const float*)d_in[5 + mod * 6 + 0];
    const float* wk = (const float*)d_in[4 + mod * 6 + 2];
    const float* bk = (const float*)d_in[5 + mod * 6 + 2];
    const float* wv = (const float*)d_in[4 + mod * 6 + 4];
    const float* bv = (const float*)d_in[5 + mod * 6 + 4];
    conv1x1_wmma<<<grd, blk, 0, stream>>>(x, wq, bq, s, mod, 1.0f / 16.0f,
        MODE_BF16_PM, 0.f, nullptr, (void*)(Q + mod * NB), C, C, HW);
    conv1x1_wmma<<<grd, blk, 0, stream>>>(x, wk, bk, s, mod, 1.0f,
        MODE_BF16_PM, 0.f, nullptr, (void*)(K + mod * NB), C, C, HW);
    conv1x1_wmma<<<grd, blk, 0, stream>>>(x, wv, bv, nullptr, -1, 1.0f,
        MODE_BF16_CM, 0.f, nullptr, (void*)(V + mod * NB), C, C, HW);
  }

  // 2) key sums (kd+kt, kr+kt, kr+kd)
  const int ndw = (int)(NB / 2);
  ksum_bf16<<<dim3(2048, 3), blk, 0, stream>>>((const unsigned*)K, (unsigned*)KS, ndw);

  // 3) per modality: flash attention -> gated-conv accumulate into FUSED
  const int qidx[3] = {0, 2, 1};   // rgb uses (q0,v0), depth (q2,v2), thermal (q1,v1)
  const float* wg = (const float*)d_in[22];
  const float* bg = (const float*)d_in[23];
  for (int m = 0; m < 3; m++) {
    flashattn_wmma<<<dim3(HW / 64, Bb), dim3(128), SMEM_BYTES, stream>>>(
        (const unsigned*)Q  + (size_t)qidx[m] * (NB / 2),
        (const unsigned*)KS + (size_t)m       * (NB / 2),
        (const unsigned*)V  + (size_t)qidx[m] * (NB / 2),
        ATT, HW);
    conv1x1_wmma<<<grd, blk, 0, stream>>>(ATT, wg, bg, nullptr, -1, 1.0f,
        MODE_GATE, (m > 0) ? 1.0f : 0.0f, ATT, (void*)FUSED, C, C, HW);
  }

  // 4) output conv into d_out
  const float* wo = (const float*)d_in[24];
  const float* bo = (const float*)d_in[25];
  conv1x1_wmma<<<grd, blk, 0, stream>>>(FUSED, wo, bo, nullptr, -1, 1.0f,
      MODE_PLAIN, 0.f, nullptr, d_out, C, OUT_C, HW);
}
